// AttentionLayer_60894046322711
// MI455X (gfx1250) — matrix operand order repaired and checked by reading, not compiled, not run on hardware
//
#include <hip/hip_runtime.h>

typedef __bf16 bf16;
typedef bf16  v4bf  __attribute__((ext_vector_type(4)));
typedef bf16  v8bf  __attribute__((ext_vector_type(8)));
typedef bf16  v16bf __attribute__((ext_vector_type(16)));
typedef float v8f   __attribute__((ext_vector_type(8)));
typedef int   v4i   __attribute__((ext_vector_type(4)));

// ---------------- CDNA5 async global->LDS staging (ASYNCcnt) ----------------
#if defined(__has_builtin)
#if __has_builtin(__builtin_amdgcn_global_load_async_to_lds_b128)
#define HAVE_ASYNC_LDS 1
#endif
#endif
#ifndef HAVE_ASYNC_LDS
#define HAVE_ASYNC_LDS 0
#endif

typedef __attribute__((address_space(1))) v4i* gv4i_t;  // global int4*
typedef __attribute__((address_space(3))) v4i* lv4i_t;  // LDS int4*

__device__ __forceinline__ void cp_b128(const bf16* g, bf16* l) {
#if HAVE_ASYNC_LDS
  __builtin_amdgcn_global_load_async_to_lds_b128((gv4i_t)g, (lv4i_t)l, 0, 0);
#else
  *(v8bf*)l = *(const v8bf*)g;
#endif
}

template <int N>
__device__ __forceinline__ void wait_async() {
#if HAVE_ASYNC_LDS
#if __has_builtin(__builtin_amdgcn_s_wait_asynccnt)
  __builtin_amdgcn_s_wait_asynccnt(N);
#else
  asm volatile("s_wait_asynccnt %0" ::"i"(N) : "memory");
#endif
#endif
}

// ---- WMMA fragment loaders (CDNA5 16x16x32 bf16 layouts, ISA 7.12.2) ----
__device__ __forceinline__ v16bf ldfragA(const bf16* row, int hl) {
  v8bf lo = *(const v8bf*)(row + 8 * hl);
  v8bf hi = *(const v8bf*)(row + 8 * hl + 16);
  return __builtin_shufflevector(lo, hi, 0,1,2,3,4,5,6,7,8,9,10,11,12,13,14,15);
}
__device__ __forceinline__ v16bf ldfragB(const bf16* row, int hl) {
  v8bf lo = *(const v8bf*)(row + 8 * hl);
  v8bf hi = *(const v8bf*)(row + 8 * hl + 16);
  return __builtin_shufflevector(lo, hi, 0,1,2,3,4,5,6,7,8,9,10,11,12,13,14,15);
}

// =====================  fp32 -> bf16 convert pass  ==========================
__global__ __launch_bounds__(256) void cvt_kernel(const float* __restrict__ in,
                                                  bf16* __restrict__ out, int n4) {
  int i = blockIdx.x * 256 + threadIdx.x;
  if (i < n4) {
    float4 v = ((const float4*)in)[i];
    v4bf o; o[0] = (bf16)v.x; o[1] = (bf16)v.y; o[2] = (bf16)v.z; o[3] = (bf16)v.w;
    ((v4bf*)out)[i] = o;
  }
}

// ==========  GEMM: out[M,N=1024] = A[M,1024] @ W[1024,1024]^T + bias  =======
// 256 threads = 8 waves (2x4); block tile 128x256; wave tile 64x64 (16 WMMA/Kstep).
// Double-buffered LDS, async global->LDS staging.
// MODE 0: out[row*1024+col].  MODE 1: store V transposed per head: [b,h,d,t].
template <typename TO, int MODE>
__global__ __launch_bounds__(256) void gemm_bias_kernel(
    const bf16* __restrict__ A, const bf16* __restrict__ W,
    const float* __restrict__ bias, TO* __restrict__ out) {
  __shared__ __align__(16) bf16 As[2][128][32];
  __shared__ __align__(16) bf16 Bs[2][256][32];

  const int tid = threadIdx.x;
  const int w = tid >> 5, l = tid & 31, hl = l >> 4, lm = l & 15;
  const int m0 = blockIdx.y * 128, n0 = blockIdx.x * 256;
  const int wm = (w >> 2) * 64, wn = (w & 3) * 64;

  auto stage = [&](int kk, int buf) {
#pragma unroll
    for (int i = 0; i < 2; ++i) {            // A tile: 512 x 16B
      int f = tid + i * 256, r = f >> 2, c = (f & 3) << 3;
      cp_b128(&A[(m0 + r) * 1024 + kk + c], &As[buf][r][c]);
    }
#pragma unroll
    for (int i = 0; i < 4; ++i) {            // W tile: 1024 x 16B
      int f = tid + i * 256, r = f >> 2, c = (f & 3) << 3;
      cp_b128(&W[(n0 + r) * 1024 + kk + c], &Bs[buf][r][c]);
    }
  };

  v8f acc[4][4] = {};
  stage(0, 0);
  for (int it = 0; it < 32; ++it) {
    const int cur = it & 1;
    if (it < 31) { stage((it + 1) * 32, cur ^ 1); wait_async<6>(); }
    else { wait_async<0>(); }
    __syncthreads();

    v16bf bfr[4];
#pragma unroll
    for (int ni = 0; ni < 4; ++ni) bfr[ni] = ldfragB(&Bs[cur][wn + ni * 16 + lm][0], hl);
#pragma unroll
    for (int mi = 0; mi < 4; ++mi) {
      v16bf a = ldfragA(&As[cur][wm + mi * 16 + lm][0], hl);
#pragma unroll
      for (int ni = 0; ni < 4; ++ni)
        acc[mi][ni] = __builtin_amdgcn_wmma_f32_16x16x32_bf16(
            false, a, false, bfr[ni], (short)0, acc[mi][ni], false, false);
    }
    __syncthreads();
  }

#pragma unroll
  for (int mi = 0; mi < 4; ++mi)
#pragma unroll
    for (int ni = 0; ni < 4; ++ni) {
      int col = n0 + wn + ni * 16 + lm;
      float bv = bias[col];
#pragma unroll
      for (int r = 0; r < 8; ++r) {
        int row = m0 + wm + mi * 16 + r + 8 * hl;
        float v = acc[mi][ni][r] + bv;
        if constexpr (MODE == 0) {
          out[row * 1024 + col] = (TO)v;
        } else {
          int b = row >> 11, t = row & 2047, h = col >> 6, d = col & 63;
          out[((b * 16 + h) * 64 + d) * 2048 + t] = (TO)v;
        }
      }
    }
}

// =====================  Flash attention (causal), bf16 WMMA  ================
// Block = 128 threads = 4 waves; 64 queries per block; V pre-transposed [b,h,d,t].
#define TSEQ 2048
__global__ __launch_bounds__(128) void attn_kernel(
    const bf16* __restrict__ Qp, const bf16* __restrict__ Kp,
    const bf16* __restrict__ Vt_g, bf16* __restrict__ Op) {
  __shared__ __align__(16) bf16 Qs[64][64];
  __shared__ __align__(16) bf16 Ks[2][64][64];
  __shared__ __align__(16) bf16 Vt[2][64][64];   // [d][key]
  __shared__ __align__(16) bf16 Ps[4][16][64];

  const int tid = threadIdx.x;
  const int w = tid >> 5, l = tid & 31, hl = l >> 4, lm = l & 15;
  int idx = blockIdx.x;
  const int qb = idx & 31; idx >>= 5;
  const int h  = idx & 15; idx >>= 4;
  const int b  = idx;
  const int rowQ0 = b * TSEQ + qb * 64;

  // stage Q once (synchronous copy)
#pragma unroll
  for (int i = 0; i < 4; ++i) {
    int f = tid + i * 128, r = f >> 3, c = (f & 7) << 3;
    *(v8bf*)&Qs[r][c] = *(const v8bf*)&Qp[(rowQ0 + r) * 1024 + h * 64 + c];
  }

  auto stageKV = [&](int kb, int buf) {
    const int rowK0 = b * TSEQ + kb * 64;
#pragma unroll
    for (int i = 0; i < 4; ++i) {      // K block: [key][d]
      int f = tid + i * 128, r = f >> 3, c = (f & 7) << 3;
      cp_b128(&Kp[(rowK0 + r) * 1024 + h * 64 + c], &Ks[buf][r][c]);
    }
#pragma unroll
    for (int i = 0; i < 4; ++i) {      // V^T block: [d][key] straight copy
      int f = tid + i * 128, d = f >> 3, c = (f & 7) << 3;
      cp_b128(&Vt_g[((b * 16 + h) * 64 + d) * 2048 + kb * 64 + c], &Vt[buf][d][c]);
    }
  };

  float m_i[8], l_i[8];
  v8f O[4] = {};
#pragma unroll
  for (int r = 0; r < 8; ++r) { m_i[r] = -1e30f; l_i[r] = 0.f; }

  stageKV(0, 0);
  for (int kb = 0; kb <= qb; ++kb) {
    const int cur = kb & 1;
    if (kb < qb) { stageKV(kb + 1, cur ^ 1); wait_async<8>(); }
    else { wait_async<0>(); }
    __syncthreads();

    // ---- S = Q @ K^T ----
    v8f S[4] = {};
#pragma unroll
    for (int kc = 0; kc < 2; ++kc) {
      v16bf aq = ldfragA(&Qs[w * 16 + lm][kc * 32], hl);
#pragma unroll
      for (int j = 0; j < 4; ++j) {
        v16bf bk = ldfragB(&Ks[cur][j * 16 + lm][kc * 32], hl);
        S[j] = __builtin_amdgcn_wmma_f32_16x16x32_bf16(
            false, aq, false, bk, (short)0, S[j], false, false);
      }
    }
    // scale + causal mask (diagonal block only)
#pragma unroll
    for (int j = 0; j < 4; ++j)
#pragma unroll
      for (int r = 0; r < 8; ++r) {
        float s = S[j][r] * 0.125f;
        if (kb == qb) {
          int kcol = j * 16 + lm, qrow = w * 16 + r + 8 * hl;
          if (kcol > qrow) s = -1e30f;
        }
        S[j][r] = s;
      }

    // ---- online softmax (row = r + 8*hl; reduce across 16-lane half) ----
#pragma unroll
    for (int r = 0; r < 8; ++r) {
      float mx = fmaxf(fmaxf(S[0][r], S[1][r]), fmaxf(S[2][r], S[3][r]));
#pragma unroll
      for (int msk = 1; msk < 16; msk <<= 1) mx = fmaxf(mx, __shfl_xor(mx, msk, 32));
      float mnew = fmaxf(m_i[r], mx);
      float alpha = __expf(m_i[r] - mnew);
      float psum = 0.f;
#pragma unroll
      for (int j = 0; j < 4; ++j) {
        float p = __expf(S[j][r] - mnew);
        S[j][r] = p;
        psum += p;
      }
#pragma unroll
      for (int msk = 1; msk < 16; msk <<= 1) psum += __shfl_xor(psum, msk, 32);
      l_i[r] = l_i[r] * alpha + psum;
      m_i[r] = mnew;
#pragma unroll
      for (int j = 0; j < 4; ++j) O[j][r] = O[j][r] * alpha;
    }

    // ---- P -> LDS (C-layout -> A-fragment layout round trip) ----
#pragma unroll
    for (int j = 0; j < 4; ++j)
#pragma unroll
      for (int r = 0; r < 8; ++r)
        Ps[w][r + 8 * hl][j * 16 + lm] = (bf16)S[j][r];

    // ---- O += P @ V ----
#pragma unroll
    for (int kc = 0; kc < 2; ++kc) {
      v16bf ap = ldfragA(&Ps[w][lm][kc * 32], hl);
#pragma unroll
      for (int j = 0; j < 4; ++j) {
        v16bf bv = ldfragB(&Vt[cur][j * 16 + lm][kc * 32], hl);
        O[j] = __builtin_amdgcn_wmma_f32_16x16x32_bf16(
            false, ap, false, bv, (short)0, O[j], false, false);
      }
    }
    __syncthreads();
  }

  // ---- epilogue: normalize, store bf16 [b, t, h*64 + d] ----
#pragma unroll
  for (int j = 0; j < 4; ++j)
#pragma unroll
    for (int r = 0; r < 8; ++r) {
      int row = rowQ0 + w * 16 + r + 8 * hl;
      int col = h * 64 + j * 16 + lm;
      Op[row * 1024 + col] = (bf16)(O[j][r] / l_i[r]);
    }
}

// ============================  launcher  ====================================
extern "C" void kernel_launch(void* const* d_in, const int* in_sizes, int n_in,
                              void* d_out, int out_size, void* d_ws, size_t ws_size,
                              hipStream_t stream) {
  const float* q  = (const float*)d_in[0];
  const float* k  = (const float*)d_in[1];
  const float* v  = (const float*)d_in[2];
  const float* Wq = (const float*)d_in[3]; const float* bq = (const float*)d_in[4];
  const float* Wk = (const float*)d_in[5]; const float* bk = (const float*)d_in[6];
  const float* Wv = (const float*)d_in[7]; const float* bv = (const float*)d_in[8];
  const float* Wo = (const float*)d_in[9]; const float* bo = (const float*)d_in[10];

  const size_t MD = (size_t)8192 * 1024;   // tokens x model-dim
  const size_t WD = (size_t)1024 * 1024;   // weight elements
  bf16* ws   = (bf16*)d_ws;
  bf16* Xbuf = ws;                 // converted input / attention output (reused)
  bf16* Wbuf = ws + MD;            // converted weight (reused)
  bf16* Qp   = ws + MD + WD;
  bf16* Kp   = Qp + MD;
  bf16* Vt   = Kp + MD;            // V projection stored transposed [b,h,d,t]

  const int nX4 = (int)(MD / 4), nW4 = (int)(WD / 4);
  dim3 cg_x((nX4 + 255) / 256), cg_w((nW4 + 255) / 256);
  dim3 gg(1024 / 256, 8192 / 128);

  cvt_kernel<<<cg_x, 256, 0, stream>>>(q, Xbuf, nX4);
  cvt_kernel<<<cg_w, 256, 0, stream>>>(Wq, Wbuf, nW4);
  gemm_bias_kernel<bf16, 0><<<gg, 256, 0, stream>>>(Xbuf, Wbuf, bq, Qp);

  cvt_kernel<<<cg_x, 256, 0, stream>>>(k, Xbuf, nX4);
  cvt_kernel<<<cg_w, 256, 0, stream>>>(Wk, Wbuf, nW4);
  gemm_bias_kernel<bf16, 0><<<gg, 256, 0, stream>>>(Xbuf, Wbuf, bk, Kp);

  cvt_kernel<<<cg_x, 256, 0, stream>>>(v, Xbuf, nX4);
  cvt_kernel<<<cg_w, 256, 0, stream>>>(Wv, Wbuf, nW4);
  gemm_bias_kernel<bf16, 1><<<gg, 256, 0, stream>>>(Xbuf, Wbuf, bv, Vt);

  attn_kernel<<<4 * 16 * 32, 128, 0, stream>>>(Qp, Kp, Vt, Xbuf);

  cvt_kernel<<<cg_w, 256, 0, stream>>>(Wo, Wbuf, nW4);
  gemm_bias_kernel<float, 0><<<gg, 256, 0, stream>>>(Xbuf, Wbuf, bo, (float*)d_out);
}
